// Seq2Opt_17532056502540
// MI455X (gfx1250) — matrix operand order repaired and checked by reading, not compiled, not run on hardware
//
#include <hip/hip_runtime.h>
#include <hip/hip_bf16.h>

// ---------------------------------------------------------------------------
// Sizes
// ---------------------------------------------------------------------------
#define BB    16
#define DIMC  640
#define LSEQ  2048
#define NHEAD 4
#define KW    7
#define KDIM  4480          // KW * DIMC
#define KT    140           // KDIM / 32
#define LPAD  2056          // LSEQ + 8 (4 zero rows each side)
#define SSFN  1470
#define DD    6590          // 2*H*DIM + SSF
#define DP    6592          // DD padded to /16 and /4
#define COUT  3200          // 640 (value) + 2560 (4 heads)

typedef __attribute__((ext_vector_type(16))) _Float16 v16h;
typedef __attribute__((ext_vector_type(8)))  _Float16 v8h;
typedef __attribute__((ext_vector_type(8)))  float    v8f;
typedef __attribute__((ext_vector_type(2)))  float    v2f;

// ---------------------------------------------------------------------------
// Pack conv weights -> f16, K index = kk*640 + ci (row-major [3200][4480]).
// Rows 0..639 = conv_v, rows 640..3199 = conv_h (already h*DIM+d major).
// ---------------------------------------------------------------------------
__global__ __launch_bounds__(256) void pack_w_kernel(
    const float* __restrict__ cvw, const float* __restrict__ chw,
    _Float16* __restrict__ Wp)
{
    int idx = blockIdx.x * 256 + threadIdx.x;
    if (idx >= COUT * KDIM) return;
    int co = idx / KDIM;
    int r  = idx % KDIM;
    int kk = r / DIMC;
    int ci = r % DIMC;
    float v = (co < DIMC)
        ? cvw[((size_t)co * DIMC + ci) * KW + kk]
        : chw[((size_t)(co - DIMC) * DIMC + ci) * KW + kk];
    Wp[idx] = (_Float16)v;
}

// ---------------------------------------------------------------------------
// Transpose emb [b][ci][l] f32 -> embT [b][l+4][ci] f16 (LDS-tiled, +halo).
// Because embT row stride == 640 == ci-block size, the K=4480 panel of any
// output column l is CONTIGUOUS in embT starting at row l+1: the conv GEMM's
// B walk is a single linear byte stream.
// ---------------------------------------------------------------------------
__global__ void transpose_kernel(const float* __restrict__ emb,
                                 _Float16* __restrict__ embT)
{
    __shared__ float tile[32][33];
    const int l0 = blockIdx.x * 32;
    const int c0 = blockIdx.y * 32;
    const int b  = blockIdx.z;
    const float* src = emb + ((size_t)b * DIMC + c0) * LSEQ + l0;
#pragma unroll
    for (int i = 0; i < 32; i += 8)
        tile[threadIdx.y + i][threadIdx.x] =
            src[(size_t)(threadIdx.y + i) * LSEQ + threadIdx.x];
    __syncthreads();
    _Float16* dst = embT + ((size_t)b * LPAD + (l0 + 4)) * DIMC + c0;
#pragma unroll
    for (int i = 0; i < 32; i += 8)
        dst[(size_t)(threadIdx.y + i) * DIMC + threadIdx.x] =
            (_Float16)tile[threadIdx.x][threadIdx.y + i];
}

__global__ void pad_kernel(_Float16* __restrict__ embT)
{
    int idx = blockIdx.x * 256 + threadIdx.x;       // 16*8*640
    if (idx >= BB * 8 * DIMC) return;
    int b = idx / (8 * DIMC);
    int r = (idx / DIMC) % 8;
    int c = idx % DIMC;
    int row = (r < 4) ? r : (LSEQ + r);
    embT[((size_t)b * LPAD + row) * DIMC + c] = (_Float16)0.f;
}

// ---------------------------------------------------------------------------
// A-fragment: 16-bit A 16x32 -> lane(m=lane&15, hi=lane>>4)
// holds A[m][hi*16..+15]: two contiguous 16B loads.
// ---------------------------------------------------------------------------
__device__ __forceinline__ v16h load_afrag(const _Float16* rowp, int k0, int hi)
{
    const v8h a0 = *(const v8h*)(rowp + k0 + hi * 16);
    const v8h a1 = *(const v8h*)(rowp + k0 + hi * 16 + 8);
    v16h r;
#pragma unroll
    for (int t = 0; t < 8; ++t) { r[t] = a0[t]; r[t + 8] = a1[t]; }
    return r;
}

__device__ __forceinline__ v8f wmma_f16(const v16h a, const v16h b, const v8f c)
{
    return __builtin_amdgcn_wmma_f32_16x16x32_f16(
        false, a, false, b, (short)0, c, false, false);
}

// ---------------------------------------------------------------------------
// Fused conv(value)+conv(scores)+softmax+sum/max.
// grid = (40 d-tiles, 16 batch), 256 threads = 8 waves = (4 heads x 2 halves).
// Each wave processes 4 N-tiles at once: per K-step 2 A-frag + 4 B-frag loads
// (192 B per lane) feed 8 WMMAs, all with linear immediate-offset addressing.
// Online softmax stats per lane, butterfly-merged across lanes, LDS-merged
// across halves.
// ---------------------------------------------------------------------------
__global__ __launch_bounds__(256) void conv_attn_kernel(
    const _Float16* __restrict__ Wp, const _Float16* __restrict__ embT,
    const float* __restrict__ bvv, const float* __restrict__ bhh,
    float* __restrict__ xsum, float* __restrict__ xmax)
{
    __shared__ float stat[NHEAD * 2 * 16 * 4];
    const int b    = blockIdx.y;
    const int d0   = blockIdx.x * 16;
    const int tid  = threadIdx.x;
    const int wave = tid >> 5;
    const int lane = tid & 31;
    const int lo   = lane & 15;
    const int hi   = lane >> 4;
    const int rb   = hi * 8;            // C/D row base for this lane
    const int h    = wave & 3;
    const int half = wave >> 2;

    const _Float16* eb = embT + (size_t)b * LPAD * DIMC;
    const _Float16* Av = Wp + (size_t)(d0 + lo) * KDIM;                    // value rows
    const _Float16* As = Wp + (size_t)(DIMC + h * DIMC + d0 + lo) * KDIM;  // score rows

    float bv[8], bs[8];
#pragma unroll
    for (int j = 0; j < 8; ++j) {
        bv[j] = bvv[d0 + rb + j];
        bs[j] = bhh[h * DIMC + d0 + rb + j];
    }

    float m8[8], S8[8], W8[8], T8[8];
#pragma unroll
    for (int j = 0; j < 8; ++j) { m8[j] = -1e30f; S8[j] = 0.f; W8[j] = 0.f; T8[j] = 0.f; }

    // online-softmax update for one 16x16 tile pair (value acc, score acc)
    auto update = [&](const v8f accv, const v8f accs) {
#pragma unroll
        for (int j = 0; j < 8; ++j) {
            const float s  = accs[j] + bs[j];
            const float v  = accv[j] + bv[j];
            const float nm = fmaxf(m8[j], s);
            const float e  = __expf(s - nm);
            const float sc = __expf(m8[j] - nm);
            const float ev = e * v;
            T8[j] = (S8[j] == 0.f) ? ev : fmaxf(T8[j] * sc, ev);
            S8[j] = S8[j] * sc + e;
            W8[j] = W8[j] * sc + ev;
            m8[j] = nm;
        }
    };

    for (int g = 0; g < 16; ++g) {
        const int l0 = (half * 64 + g * 4) * 16;   // first of 4 consecutive tiles
        // linear B stream base for this lane's column (tile q at +q*16*640)
        const _Float16* bbase = eb + (size_t)(l0 + lo + 1) * DIMC + hi * 16;
        v8f av0 = {}, av1 = {}, av2 = {}, av3 = {};
        v8f as0 = {}, as1 = {}, as2 = {}, as3 = {};
        for (int kt = 0; kt < KT; ++kt) {
            const int ko = kt * 32;
            const v16h af = load_afrag(Av, ko, hi);
            const v16h ag = load_afrag(As, ko, hi);
            const v16h b0 = *(const v16h*)(bbase + ko);
            const v16h b1 = *(const v16h*)(bbase + ko + 16 * DIMC);
            const v16h b2 = *(const v16h*)(bbase + ko + 32 * DIMC);
            const v16h b3 = *(const v16h*)(bbase + ko + 48 * DIMC);
            __builtin_prefetch(bbase + ko + 128, 0, 3);   // global_prefetch, near caches
            av0 = wmma_f16(af, b0, av0);
            as0 = wmma_f16(ag, b0, as0);
            av1 = wmma_f16(af, b1, av1);
            as1 = wmma_f16(ag, b1, as1);
            av2 = wmma_f16(af, b2, av2);
            as2 = wmma_f16(ag, b2, as2);
            av3 = wmma_f16(af, b3, av3);
            as3 = wmma_f16(ag, b3, as3);
        }
        update(av0, as0);
        update(av1, as1);
        update(av2, as2);
        update(av3, as3);
    }

    // butterfly merge across the 16 lanes of each half-wave
#pragma unroll
    for (int off = 1; off < 16; off <<= 1) {
#pragma unroll
        for (int j = 0; j < 8; ++j) {
            const float om = __shfl_xor(m8[j], off, 32);
            const float oS = __shfl_xor(S8[j], off, 32);
            const float oW = __shfl_xor(W8[j], off, 32);
            const float oT = __shfl_xor(T8[j], off, 32);
            const float nm = fmaxf(m8[j], om);
            const float s1 = __expf(m8[j] - nm);
            const float s2 = __expf(om - nm);
            S8[j] = S8[j] * s1 + oS * s2;
            W8[j] = W8[j] * s1 + oW * s2;
            T8[j] = fmaxf(T8[j] * s1, oT * s2);
            m8[j] = nm;
        }
    }

    if (lo == 0) {
#pragma unroll
        for (int j = 0; j < 8; ++j) {
            float* p = &stat[(((h * 2 + half) * 16) + rb + j) * 4];
            p[0] = m8[j]; p[1] = S8[j]; p[2] = W8[j]; p[3] = T8[j];
        }
    }
    __syncthreads();
    if (tid < 64) {
        const int hh = tid >> 4, row = tid & 15;
        const float* p0 = &stat[((hh * 2 + 0) * 16 + row) * 4];
        const float* p1 = &stat[((hh * 2 + 1) * 16 + row) * 4];
        const float nm = fmaxf(p0[0], p1[0]);
        const float s1 = __expf(p0[0] - nm);
        const float s2 = __expf(p1[0] - nm);
        const float S  = p0[1] * s1 + p1[1] * s2;
        const float Wv = p0[2] * s1 + p1[2] * s2;
        const float Tv = fmaxf(p0[3] * s1, p1[3] * s2);
        const float inv = 1.f / S;
        const int g = hh * DIMC + d0 + row;
        xsum[b * 2560 + g] = Wv * inv;
        xmax[b * 2560 + g] = Tv * inv;
    }
}

// ---------------------------------------------------------------------------
// cat([x_sum, x_max, ssf]) + top-level BN -> xa [16][6592] (padded, zeros).
// ---------------------------------------------------------------------------
__global__ __launch_bounds__(256) void bn_cat_kernel(
    const float* __restrict__ xsum, const float* __restrict__ xmax,
    const float* __restrict__ ssf,
    const float* __restrict__ g, const float* __restrict__ be,
    const float* __restrict__ mn, const float* __restrict__ vr,
    float* __restrict__ xa)
{
    int idx = blockIdx.x * 256 + threadIdx.x;
    if (idx >= BB * DP) return;
    int b = idx / DP, i = idx % DP;
    float o = 0.f;
    if (i < DD) {
        float val = (i < 2560) ? xsum[b * 2560 + i]
                  : (i < 5120) ? xmax[b * 2560 + i - 2560]
                               : ssf[b * SSFN + i - 5120];
        o = (val - mn[i]) * (g[i] * rsqrtf(vr[i] + 1e-5f)) + be[i];
    }
    xa[(size_t)b * DP + i] = o;
}

// ---------------------------------------------------------------------------
// RD block: h = x @ W^T + b; BN; LeakyReLU(0.01); x += h.
// M = 16 (batch) exactly one WMMA tile -> native f32 WMMA 16x16x4.
// A = x rows, B = rd_w rows: both 8B contiguous loads, no repacking.
// ---------------------------------------------------------------------------
__global__ __launch_bounds__(128) void rd_kernel(
    const float* __restrict__ xin, float* __restrict__ xout,
    const float* __restrict__ Wl, const float* __restrict__ bl,
    const float* __restrict__ bg, const float* __restrict__ bb2,
    const float* __restrict__ bm, const float* __restrict__ bvr)
{
    const int tid  = threadIdx.x;
    const int wave = tid >> 5;
    const int lane = tid & 31;
    const int lo   = lane & 15;
    const int hi   = lane >> 4;
    const int rb   = hi * 8;
    const int nt   = blockIdx.x * 4 + wave;          // 0..411
    const int iw   = nt * 16 + lo;                    // output column
    const int ic   = (iw < DD) ? iw : (DD - 1);       // clamp for loads only
    const float* wr = Wl + (size_t)ic * DD;           // B row (rd_w row ic)
    const float* xr = xin + (size_t)lo * DP;          // A row (batch lo)

    v8f acc = {};
#if __has_builtin(__builtin_amdgcn_wmma_f32_16x16x4_f32)
    for (int k0 = 0; k0 < 6588; k0 += 4) {
        const v2f a  = *(const v2f*)(xr + k0 + 2 * hi);
        const v2f bf = *(const v2f*)(wr + k0 + 2 * hi);
        acc = __builtin_amdgcn_wmma_f32_16x16x4_f32(
            false, a, false, bf, (short)0, acc, false, false);
    }
    // K tail (k = 6588, 6589) done scalar to stay inside rd_w rows
#pragma unroll
    for (int j = 0; j < 8; ++j) {
        acc[j] += xin[(size_t)(rb + j) * DP + 6588] * wr[6588]
                + xin[(size_t)(rb + j) * DP + 6589] * wr[6589];
    }
#else
    for (int k = 0; k < DD; ++k) {
        const float w = wr[k];
#pragma unroll
        for (int j = 0; j < 8; ++j)
            acc[j] += xin[(size_t)(rb + j) * DP + k] * w;
    }
#endif

    if (iw < DD) {
        const float scale = bg[iw] * rsqrtf(bvr[iw] + 1e-5f);
        const float bbv = bb2[iw], bmv = bm[iw], blv = bl[iw];
#pragma unroll
        for (int j = 0; j < 8; ++j) {
            float hsum = acc[j] + blv;
            float hb   = (hsum - bmv) * scale + bbv;
            hb = (hb > 0.f) ? hb : 0.01f * hb;
            xout[(size_t)(rb + j) * DP + iw] =
                xin[(size_t)(rb + j) * DP + iw] + hb;
        }
    } else {
#pragma unroll
        for (int j = 0; j < 8; ++j)
            xout[(size_t)(rb + j) * DP + iw] = 0.f;   // keep K-padding zero
    }
}

// ---------------------------------------------------------------------------
// Output head: out[b] = x[b] . out_w + out_b
// ---------------------------------------------------------------------------
__global__ __launch_bounds__(256) void out_head_kernel(
    const float* __restrict__ x, const float* __restrict__ ow,
    const float* __restrict__ ob, float* __restrict__ out)
{
    __shared__ float red[256];
    const int tid = threadIdx.x;
    float part[BB];
#pragma unroll
    for (int b = 0; b < BB; ++b) part[b] = 0.f;
    for (int i = tid; i < DD; i += 256) {
        const float w = ow[i];
#pragma unroll
        for (int b = 0; b < BB; ++b) part[b] += x[(size_t)b * DP + i] * w;
    }
    for (int b = 0; b < BB; ++b) {
        red[tid] = part[b];
        __syncthreads();
        for (int s = 128; s > 0; s >>= 1) {
            if (tid < s) red[tid] += red[tid + s];
            __syncthreads();
        }
        if (tid == 0) out[b] = red[0] + ob[0];
        __syncthreads();
    }
}

// ---------------------------------------------------------------------------
// Launch
// ---------------------------------------------------------------------------
extern "C" void kernel_launch(void* const* d_in, const int* in_sizes, int n_in,
                              void* d_out, int out_size, void* d_ws, size_t ws_size,
                              hipStream_t stream)
{
    (void)in_sizes; (void)n_in; (void)out_size; (void)ws_size;
    const float* emb = (const float*)d_in[0];
    const float* ssf = (const float*)d_in[1];
    const float* cvw = (const float*)d_in[2];
    const float* cvb = (const float*)d_in[3];
    const float* chw = (const float*)d_in[4];
    const float* chb = (const float*)d_in[5];
    const float* bng = (const float*)d_in[6];
    const float* bnb = (const float*)d_in[7];
    const float* bnm = (const float*)d_in[8];
    const float* bnv = (const float*)d_in[9];
    const float* rdw = (const float*)d_in[10];
    const float* rdb = (const float*)d_in[11];
    const float* rbg = (const float*)d_in[12];
    const float* rbb = (const float*)d_in[13];
    const float* rbm = (const float*)d_in[14];
    const float* rbv = (const float*)d_in[15];
    const float* ow  = (const float*)d_in[16];
    const float* ob  = (const float*)d_in[17];
    float* out = (float*)d_out;

    // workspace layout (all 256B aligned), total ~68.7 MB
    char* ws = (char*)d_ws;
    _Float16* Wp   = (_Float16*)(ws + 0);                    // 3200*4480*2 = 28,672,000
    _Float16* embT = (_Float16*)(ws + 28672000ull);          // 16*2056*640*2 = 42,106,880
    float*    xsum = (float*)(ws + 70778880ull);             // 16*2560*4
    float*    xmax = (float*)(ws + 70942720ull);             // 16*2560*4
    float*    xa   = (float*)(ws + 71106560ull);             // 16*6592*4
    float*    xb   = (float*)(ws + 71528448ull);             // 16*6592*4

    pack_w_kernel<<<56000, 256, 0, stream>>>(cvw, chw, Wp);
    transpose_kernel<<<dim3(LSEQ / 32, DIMC / 32, BB), dim3(32, 8), 0, stream>>>(emb, embT);
    pad_kernel<<<(BB * 8 * DIMC + 255) / 256, 256, 0, stream>>>(embT);

    conv_attn_kernel<<<dim3(DIMC / 16, BB), 256, 0, stream>>>(Wp, embT, cvb, chb, xsum, xmax);

    bn_cat_kernel<<<(BB * DP + 255) / 256, 256, 0, stream>>>(
        xsum, xmax, ssf, bng, bnb, bnm, bnv, xa);

    const size_t WS2 = (size_t)DD * DD;
    rd_kernel<<<103, 128, 0, stream>>>(xa, xb, rdw,           rdb,            rbg,            rbb,            rbm,            rbv);
    rd_kernel<<<103, 128, 0, stream>>>(xb, xa, rdw + WS2,     rdb + DD,       rbg + DD,       rbb + DD,       rbm + DD,       rbv + DD);
    rd_kernel<<<103, 128, 0, stream>>>(xa, xb, rdw + 2 * WS2, rdb + 2 * DD,   rbg + 2 * DD,   rbb + 2 * DD,   rbm + 2 * DD,   rbv + 2 * DD);

    out_head_kernel<<<1, 256, 0, stream>>>(xb, ow, ob, out);
}
